// PronunciationPredictor_7550552506779
// MI455X (gfx1250) — matrix operand order repaired and checked by reading, not compiled, not run on hardware
//
#include <hip/hip_runtime.h>
#include <math.h>

// ---------------------------------------------------------------------------
// CDNA5 (gfx1250): bf16 WMMA GEMMs with async global->LDS DMA staging.
//  - all weights/activations pre-converted to padded bf16 (K->mult64, rows->mult128)
//  - GEMM inner loop: double-buffered 128x64 LDS slabs staged with
//    global_load_async_to_lds_b128 + s_wait_asynccnt pipelining,
//    16x v_wmma_f32_16x16x32_bf16 per synchronization.
// ---------------------------------------------------------------------------

typedef __attribute__((ext_vector_type(16))) __bf16 bf16x16;
typedef __attribute__((ext_vector_type(8)))  float  floatx8;

#define TPB 256
#define BM 128
#define BN 128
#define BKS 64          // staged K per slab (2 WMMA k-steps of 32)

static inline int rup(int x, int a) { return (x + a - 1) / a * a; }

// ------------------------- ragged segment mean ------------------------------
__global__ __launch_bounds__(TPB) void seg_mean_kernel(
    const float* __restrict__ wav, const int* __restrict__ start,
    const int* __restrict__ end, float* __restrict__ out, int T, int D, int Wn)
{
  int bw = blockIdx.x;               // b*Wn + w
  int b  = bw / Wn;
  int sf = start[bw] / 320; sf = sf < 0 ? 0 : (sf > T ? T : sf);
  int ef = end[bw]   / 320; ef = ef < 0 ? 0 : (ef > T ? T : ef);
  int len = ef - sf;
  float inv = len > 0 ? 1.0f / (float)len : 0.0f;
  const float* base = wav + (size_t)b * T * D;
  float* op = out + (size_t)bw * D;
  for (int d = threadIdx.x; d < D; d += TPB) {
    float s = 0.f;
    for (int t = sf; t < ef; ++t) s += base[(size_t)t * D + d];
    op[d] = s * inv;
  }
}

// ------------------------- column-slice copy (concat) -----------------------
__global__ __launch_bounds__(TPB) void copy_cols_kernel(
    const float* __restrict__ src, float* __restrict__ dst,
    int rows, int cols, int src_ld, int dst_ld, int dst_off)
{
  size_t total = (size_t)rows * cols;
  for (size_t i = (size_t)blockIdx.x * TPB + threadIdx.x; i < total;
       i += (size_t)gridDim.x * TPB) {
    size_t r = i / cols; int c = (int)(i % cols);
    dst[r * (size_t)dst_ld + dst_off + c] = src[r * (size_t)src_ld + c];
  }
}

// ------------------------- mean over axis 1 ---------------------------------
__global__ __launch_bounds__(TPB) void mean_axis1_kernel(
    const float* __restrict__ in, float* __restrict__ out,
    int A, int L, int D, int out_ld, int out_off)
{
  int a = blockIdx.y;
  int d = blockIdx.x * TPB + threadIdx.x;
  if (d >= D) return;
  const float* p = in + (size_t)a * L * D + d;
  float s = 0.f;
  for (int l = 0; l < L; ++l) s += p[(size_t)l * D];
  out[(size_t)a * out_ld + out_off + d] = s / (float)L;
}

// ------------------- fp32 -> padded bf16 conversion -------------------------
// dst is [Rp, Kp] row-major bf16, zero-filled outside [R, K].  Kp % 8 == 0.
__global__ __launch_bounds__(TPB) void cvt_pad_kernel(
    const float* __restrict__ src, __bf16* __restrict__ dst,
    int R, int K, int Rp, int Kp)
{
  size_t total8 = ((size_t)Rp * Kp) >> 3;        // 8 elems (16B) per step
  for (size_t i = (size_t)blockIdx.x * TPB + threadIdx.x; i < total8;
       i += (size_t)gridDim.x * TPB) {
    size_t e0 = i << 3;
    size_t r  = e0 / Kp;
    int   k0  = (int)(e0 % Kp);                  // whole 8-chunk in one row
    __bf16 tmp[8];
    bool rowok = (int)r < R;
    const float* s = src + r * (size_t)K + k0;
    if (rowok && k0 + 8 <= K) {
      const float4* s4 = (const float4*)s;
      float4 a = s4[0], b = s4[1];
      tmp[0] = (__bf16)a.x; tmp[1] = (__bf16)a.y; tmp[2] = (__bf16)a.z; tmp[3] = (__bf16)a.w;
      tmp[4] = (__bf16)b.x; tmp[5] = (__bf16)b.y; tmp[6] = (__bf16)b.z; tmp[7] = (__bf16)b.w;
    } else {
#pragma unroll
      for (int j = 0; j < 8; ++j) {
        int k = k0 + j;
        tmp[j] = (rowok && k < K) ? (__bf16)s[j] : (__bf16)0.0f;
      }
    }
    *(uint4*)(dst + e0) = *(const uint4*)tmp;    // 16B aligned store
  }
}

// ------------------------- WMMA fragment loaders ----------------------------
// 16-bit A-matrix 16x32 (MxK), wave32 layout (ISA 7.12.2):
//   lane<16: M=lane,    K in {0..7, 16..23}; lane>=16: M=lane-16, K in {8..15, 24..31}
__device__ __forceinline__ bf16x16 load_a_frag(const __bf16* __restrict__ t, int ld, int lane)
{
  int l = lane & 15, hi = (lane >> 4) & 1;
  const __bf16* q = t + l * ld + hi * 8;
  bf16x16 r;
#pragma unroll
  for (int i = 0; i < 8; ++i) { r[i] = q[i]; r[i + 8] = q[16 + i]; }
  return r;
}
// 16-bit B-matrix 32x16 (KxN): lane&15 -> N, K in {0..7,16..23} (lane<16) / {8..15,24..31} (lane>=16).
// Bs slab stored [n][k] row-major.
__device__ __forceinline__ bf16x16 load_b_frag(const __bf16* __restrict__ t, int ld, int lane)
{
  int l = lane & 15, hi = (lane >> 4) & 1;
  const __bf16* q = t + l * ld + hi * 8;
  bf16x16 r;
#pragma unroll
  for (int i = 0; i < 8; ++i) { r[i] = q[i]; r[i + 8] = q[16 + i]; }
  return r;
}

// -------- async DMA of one 128x64 bf16 slab (16KB) into LDS -----------------
// 1024 16-byte chunks, 256 threads -> 4 global_load_async_to_lds_b128 each.
// chunk c: row = c>>3, oct = c&7 ; global = g + row*ldk + oct*8 (bf16 elems),
// LDS byte offset = c*16 (slab is [row][64] bf16 row-major = row*128B).
__device__ __forceinline__ void async_tile(const __bf16* __restrict__ g, int ldk,
                                           const __bf16* lds, int tid)
{
#pragma unroll
  for (int j = 0; j < 4; ++j) {
    int c = tid + j * 256;
    int row = c >> 3, q = c & 7;
    unsigned loff = (unsigned)(size_t)lds + (unsigned)c * 16u;
    unsigned long long gaddr =
        (unsigned long long)(size_t)(g + (size_t)row * ldk + q * 8);
    asm volatile("global_load_async_to_lds_b128 %0, %1, off"
                 :: "v"(loff), "v"(gaddr) : "memory");
  }
}

// ------------------------- bf16 WMMA GEMM  C = A*W^T + bias -----------------
// A: [M (mult 128), Kp (mult 64)] bf16 (zero-padded),
// W: [Np >= N (mult 128), Kp] bf16 (zero-padded rows), C: [M, N] fp32.
__global__ __launch_bounds__(TPB) void gemm_bf16_wmma(
    const __bf16* __restrict__ A, const __bf16* __restrict__ W,
    const float* __restrict__ bias, float* __restrict__ C,
    int M, int N, int Kp, int relu)
{
  __shared__ __align__(16) __bf16 As[2][BM * BKS];
  __shared__ __align__(16) __bf16 Bs[2][BN * BKS];

  int tid  = threadIdx.x;
  int lane = tid & 31;
  int wave = tid >> 5;
  int m0 = blockIdx.y * BM;
  int n0 = blockIdx.x * BN;
  int wm = (wave >> 2) * 64;
  int wn = (wave & 3) * 32;

  floatx8 acc[4][2];
#pragma unroll
  for (int i = 0; i < 4; ++i)
#pragma unroll
    for (int j = 0; j < 2; ++j) acc[i][j] = (floatx8)0.0f;

  const __bf16* Abase = A + (size_t)m0 * Kp;
  const __bf16* Wbase = W + (size_t)n0 * Kp;
  int nk = Kp / BKS;

  // prologue: stage slab 0 into buffer 0
  async_tile(Abase, Kp, &As[0][0], tid);
  async_tile(Wbase, Kp, &Bs[0][0], tid);

  for (int i = 0; i < nk; ++i) {
    int cur = i & 1, nxt = cur ^ 1;
    if (i + 1 < nk) {
      // stage next slab, then wait until only those 8 asyncs are outstanding
      async_tile(Abase + (size_t)(i + 1) * BKS, Kp, &As[nxt][0], tid);
      async_tile(Wbase + (size_t)(i + 1) * BKS, Kp, &Bs[nxt][0], tid);
      asm volatile("s_wait_asynccnt 0x8" ::: "memory");
    } else {
      asm volatile("s_wait_asynccnt 0x0" ::: "memory");
    }
    __syncthreads();   // all waves' slab-i data visible in LDS

    const __bf16* Ac = &As[cur][0];
    const __bf16* Bc = &Bs[cur][0];
#pragma unroll
    for (int ks = 0; ks < BKS; ks += 32) {
      bf16x16 afr[4], bfr[2];
#pragma unroll
      for (int a = 0; a < 4; ++a)
        afr[a] = load_a_frag(Ac + (wm + a * 16) * BKS + ks, BKS, lane);
#pragma unroll
      for (int b = 0; b < 2; ++b)
        bfr[b] = load_b_frag(Bc + (wn + b * 16) * BKS + ks, BKS, lane);
#pragma unroll
      for (int a = 0; a < 4; ++a)
#pragma unroll
        for (int b = 0; b < 2; ++b)
          acc[a][b] = __builtin_amdgcn_wmma_f32_16x16x32_bf16(
              false, afr[a], false, bfr[b], (short)0, acc[a][b], false, false);
    }
    __syncthreads();   // reads done before buffer is overwritten next iter
  }

  // ---- store C (16x16 f32 C/D layout: VGPR r -> M=r(+8), lane -> N) ----
  int ln = lane & 15;
  int hi = lane >> 4;
#pragma unroll
  for (int i = 0; i < 4; ++i) {
#pragma unroll
    for (int j = 0; j < 2; ++j) {
      int col = n0 + wn + j * 16 + ln;
      if (col < N) {
        float bv = bias ? bias[col] : 0.0f;
#pragma unroll
        for (int r = 0; r < 8; ++r) {
          int m = m0 + wm + i * 16 + hi * 8 + r;
          if (m < M) {
            float v = acc[i][j][r] + bv;
            if (relu) v = fmaxf(v, 0.0f);
            C[(size_t)m * N + col] = v;
          }
        }
      }
    }
  }
}

// ------------------------- attention (S=16 sequence axis) -------------------
__global__ __launch_bounds__(TPB) void attn_kernel(
    const float* __restrict__ qkv, float* __restrict__ o,
    int S, int Nn, int D, int nh)
{
  __shared__ float sc[256];
  __shared__ float att[256];
  int n = blockIdx.x, h = blockIdx.y;
  int hd = D / nh;
  int tid = threadIdx.x;
  size_t ld = (size_t)3 * D;
  float scale = rsqrtf((float)hd);

  if (tid < S * S) {
    int s = tid / S, t = tid % S;
    const float* q = qkv + ((size_t)s * Nn + n) * ld + (size_t)h * hd;
    const float* k = qkv + ((size_t)t * Nn + n) * ld + D + (size_t)h * hd;
    float a = 0.f;
    for (int d = 0; d < hd; ++d) a += q[d] * k[d];
    sc[tid] = a * scale;
  }
  __syncthreads();
  if (tid < S * S) {
    int s = tid / S;
    float mx = -1e30f;
    for (int t = 0; t < S; ++t) mx = fmaxf(mx, sc[s * S + t]);
    float sum = 0.f;
    for (int t = 0; t < S; ++t) sum += expf(sc[s * S + t] - mx);
    att[tid] = expf(sc[tid] - mx) / sum;
  }
  __syncthreads();
  for (int s = 0; s < S; ++s) {
    for (int d = tid; d < hd; d += TPB) {
      float a = 0.f;
      for (int t = 0; t < S; ++t)
        a += att[s * S + t] * qkv[((size_t)t * Nn + n) * ld + 2 * (size_t)D + (size_t)h * hd + d];
      o[((size_t)s * Nn + n) * (size_t)D + (size_t)h * hd + d] = a;
    }
  }
}

// ------------------------- fused residual + LayerNorm -----------------------
__global__ __launch_bounds__(TPB) void add_ln_kernel(
    const float* __restrict__ x, const float* __restrict__ r,
    const float* __restrict__ w, const float* __restrict__ b,
    float* __restrict__ out, int D)
{
  __shared__ float s1[TPB], s2[TPB];
  size_t row = blockIdx.x;
  const float* xp = x + row * D;
  const float* rp = r + row * D;
  float* op = out + row * D;
  int tid = threadIdx.x;
  float a = 0.f, q = 0.f;
  for (int d = tid; d < D; d += TPB) {
    float v = xp[d] + rp[d];
    a += v; q += v * v;
  }
  s1[tid] = a; s2[tid] = q;
  __syncthreads();
  for (int st = TPB / 2; st > 0; st >>= 1) {
    if (tid < st) { s1[tid] += s1[tid + st]; s2[tid] += s2[tid + st]; }
    __syncthreads();
  }
  float mean = s1[0] / (float)D;
  float var  = s2[0] / (float)D - mean * mean;
  float inv  = rsqrtf(var + 1e-5f);
  for (int d = tid; d < D; d += TPB) {
    float v = xp[d] + rp[d];
    op[d] = (v - mean) * inv * w[d] + b[d];
  }
}

// ------------------------- heads --------------------------------------------
__global__ __launch_bounds__(TPB) void sen_head_kernel(
    const float* __restrict__ uttr, const float* __restrict__ w,
    const float* __restrict__ bias, float* __restrict__ out)
{
  __shared__ float red[TPB];
  int b = blockIdx.x >> 2, o = blockIdx.x & 3;
  const float* u  = uttr + (size_t)b * 3840;
  const float* wp = w + (size_t)o * 3840;
  float s = 0.f;
  for (int d = threadIdx.x; d < 3840; d += TPB) s += u[d] * wp[d];
  red[threadIdx.x] = s; __syncthreads();
  for (int st = TPB / 2; st > 0; st >>= 1) {
    if (threadIdx.x < st) red[threadIdx.x] += red[threadIdx.x + st];
    __syncthreads();
  }
  if (threadIdx.x == 0) out[o * 16 + b] = red[0] + bias[o];
}

__global__ __launch_bounds__(TPB) void word_head_kernel(
    const float* __restrict__ x, const float* __restrict__ w,
    const float* __restrict__ bias, float* __restrict__ out, int D, int rows)
{
  __shared__ float red[TPB];
  int row = blockIdx.x;
  const float* xp = x + (size_t)row * D;
  for (int o = 0; o < 3; ++o) {
    const float* wp = w + (size_t)o * D;
    float s = 0.f;
    for (int d = threadIdx.x; d < D; d += TPB) s += xp[d] * wp[d];
    red[threadIdx.x] = s; __syncthreads();
    for (int st = TPB / 2; st > 0; st >>= 1) {
      if (threadIdx.x < st) red[threadIdx.x] += red[threadIdx.x + st];
      __syncthreads();
    }
    if (threadIdx.x == 0) out[(size_t)o * rows + row] = red[0] + bias[o];
    __syncthreads();
  }
}

// ------------------------- host helpers -------------------------------------
static inline int grid_for(size_t total) {
  size_t g = (total + TPB - 1) / TPB;
  return (int)(g > 4096 ? 4096 : g);
}

// One TransformerEncoderLayer (post-norm, relu, dff=2048), rows = 2048.
static void run_enc(float* x, int D, int nh, const float* const* p,
                    const __bf16* wq, const __bf16* wo,
                    const __bf16* w1, const __bf16* w2, int KpD,
                    __bf16* abf, float* qkv, float* obuf, float* tmp, float* ffh,
                    hipStream_t st)
{
  const int rows = 2048;
  dim3 blk(TPB);
  // QKV projection
  cvt_pad_kernel<<<grid_for(((size_t)rows * KpD) >> 3), blk, 0, st>>>(x, abf, rows, D, rows, KpD);
  {
    dim3 g((3 * D + BN - 1) / BN, rows / BM);
    gemm_bf16_wmma<<<g, blk, 0, st>>>(abf, wq, p[1], qkv, rows, 3 * D, KpD, 0);
  }
  // attention over S=16
  {
    dim3 g(128, nh);
    attn_kernel<<<g, blk, 0, st>>>(qkv, obuf, 16, 128, D, nh);
  }
  // output projection
  cvt_pad_kernel<<<grid_for(((size_t)rows * KpD) >> 3), blk, 0, st>>>(obuf, abf, rows, D, rows, KpD);
  {
    dim3 g((D + BN - 1) / BN, rows / BM);
    gemm_bf16_wmma<<<g, blk, 0, st>>>(abf, wo, p[3], tmp, rows, D, KpD, 0);
  }
  add_ln_kernel<<<rows, blk, 0, st>>>(x, tmp, p[4], p[5], x, D);
  // FF1 (relu)
  cvt_pad_kernel<<<grid_for(((size_t)rows * KpD) >> 3), blk, 0, st>>>(x, abf, rows, D, rows, KpD);
  {
    dim3 g((2048 + BN - 1) / BN, rows / BM);
    gemm_bf16_wmma<<<g, blk, 0, st>>>(abf, w1, p[7], ffh, rows, 2048, KpD, 1);
  }
  // FF2
  cvt_pad_kernel<<<grid_for(((size_t)rows * 2048) >> 3), blk, 0, st>>>(ffh, abf, rows, 2048, rows, 2048);
  {
    dim3 g((D + BN - 1) / BN, rows / BM);
    gemm_bf16_wmma<<<g, blk, 0, st>>>(abf, w2, p[9], tmp, rows, D, 2048, 0);
  }
  add_ln_kernel<<<rows, blk, 0, st>>>(x, tmp, p[10], p[11], x, D);
}

// ---------------------------------------------------------------------------
extern "C" void kernel_launch(void* const* d_in, const int* in_sizes, int n_in,
                              void* d_out, int out_size, void* d_ws, size_t ws_size,
                              hipStream_t stream)
{
  (void)in_sizes; (void)n_in; (void)out_size; (void)ws_size;
  const int Bv = 16, Tv = 2000, Wn = 128, Stxt = 200;
  const int rows = Bv * Wn;                         // 2048
  const int SSL = 768, TXT = 768, ALN = 10, PHN = 142;
  const int DF = 2304, DW = 2456;

  const float* wav  = (const float*)d_in[0];
  const float* asr  = (const float*)d_in[1];
  const float* gts  = (const float*)d_in[2];
  const float* gte  = (const float*)d_in[3];
  const float* alf  = (const float*)d_in[4];
  const float* phv  = (const float*)d_in[5];
  const int*   st_  = (const int*)d_in[6];
  const int*   en_  = (const int*)d_in[7];
  const float* const* fusion_p = (const float* const*)&d_in[8];
  const float* const* word_p   = (const float* const*)&d_in[20];
  const float* const* align_p  = (const float* const*)&d_in[32];
  const float* const* phone_p  = (const float* const*)&d_in[44];
  const float* sen_w = (const float*)d_in[56];
  const float* sen_b = (const float*)d_in[57];
  const float* wrd_w = (const float*)d_in[58];
  const float* wrd_b = (const float*)d_in[59];
  float* out = (float*)d_out;

  // -------- workspace carve-out (byte based, 256B aligned blocks) --------
  char* base = (char*)d_ws;
  size_t off = 0;
  auto allocb = [&](size_t bytes) {
    char* p = base + off;
    off = (off + bytes + 255) & ~(size_t)255;
    return p;
  };
  auto allocf  = [&](size_t n) { return (float*)allocb(n * 4); };
  auto allocbf = [&](size_t n) { return (__bf16*)allocb(n * 2); };

  float* wav_al   = allocf((size_t)rows * SSL);
  float* x_fusion = allocf((size_t)rows * DF);
  float* x_word   = allocf((size_t)rows * DW);
  float* af       = allocf((size_t)rows * ALN);
  float* pv       = allocf((size_t)rows * PHN);
  float* qkv      = allocf((size_t)rows * 3 * DW);
  float* obuf     = allocf((size_t)rows * DW);
  float* tmp      = allocf((size_t)rows * DW);
  float* ffh      = allocf((size_t)rows * 2048);
  float* uttr     = allocf((size_t)Bv * 3840);

  // bf16 activation scratch: rows x max(Kp) = 2048 x 2496
  __bf16* abf = allocbf((size_t)rows * 2496);

  dim3 blk(TPB);

  // bf16 padded weights + conversion (rows -> mult 128, K -> mult 64)
  auto cvtw = [&](const float* src, int R, int K) {
    int Rp = rup(R, 128), Kp = rup(K, BKS);
    __bf16* dst = allocbf((size_t)Rp * Kp);
    cvt_pad_kernel<<<grid_for(((size_t)Rp * Kp) >> 3), blk, 0, stream>>>(src, dst, R, K, Rp, Kp);
    return dst;
  };
  // fusion (D=2304)
  const __bf16* fu_q = cvtw(fusion_p[0], 3 * DF, DF);
  const __bf16* fu_o = cvtw(fusion_p[2], DF, DF);
  const __bf16* fu_1 = cvtw(fusion_p[6], 2048, DF);
  const __bf16* fu_2 = cvtw(fusion_p[8], DF, 2048);
  // word (D=2456)
  const __bf16* wo_q = cvtw(word_p[0], 3 * DW, DW);
  const __bf16* wo_o = cvtw(word_p[2], DW, DW);
  const __bf16* wo_1 = cvtw(word_p[6], 2048, DW);
  const __bf16* wo_2 = cvtw(word_p[8], DW, 2048);
  // align (D=10)
  const __bf16* al_q = cvtw(align_p[0], 3 * ALN, ALN);
  const __bf16* al_o = cvtw(align_p[2], ALN, ALN);
  const __bf16* al_1 = cvtw(align_p[6], 2048, ALN);
  const __bf16* al_2 = cvtw(align_p[8], ALN, 2048);
  // phone (D=142)
  const __bf16* ph_q = cvtw(phone_p[0], 3 * PHN, PHN);
  const __bf16* ph_o = cvtw(phone_p[2], PHN, PHN);
  const __bf16* ph_1 = cvtw(phone_p[6], 2048, PHN);
  const __bf16* ph_2 = cvtw(phone_p[8], PHN, 2048);

  // 1) ragged segment mean -> wav_aligned [rows, 768]
  seg_mean_kernel<<<rows, blk, 0, stream>>>(wav, st_, en_, wav_al, Tv, SSL, Wn);

  // 2) align encoder (D=10, 2 heads)
  copy_cols_kernel<<<grid_for((size_t)rows * ALN), blk, 0, stream>>>(
      alf, af, rows, ALN, ALN, ALN, 0);
  run_enc(af, ALN, 2, align_p, al_q, al_o, al_1, al_2, rup(ALN, BKS),
          abf, qkv, obuf, tmp, ffh, stream);

  // 3) phone encoder (D=142, 2 heads)
  copy_cols_kernel<<<grid_for((size_t)rows * PHN), blk, 0, stream>>>(
      phv, pv, rows, PHN, PHN, PHN, 0);
  run_enc(pv, PHN, 2, phone_p, ph_q, ph_o, ph_1, ph_2, rup(PHN, BKS),
          abf, qkv, obuf, tmp, ffh, stream);

  // 4) fusion encoder (D=2304, 8 heads): concat(wav_al, gte, asr)
  copy_cols_kernel<<<grid_for((size_t)rows * SSL), blk, 0, stream>>>(
      wav_al, x_fusion, rows, SSL, SSL, DF, 0);
  copy_cols_kernel<<<grid_for((size_t)rows * TXT), blk, 0, stream>>>(
      gte, x_fusion, rows, TXT, TXT, DF, 768);
  copy_cols_kernel<<<grid_for((size_t)rows * TXT), blk, 0, stream>>>(
      asr, x_fusion, rows, TXT, TXT, DF, 1536);
  run_enc(x_fusion, DF, 8, fusion_p, fu_q, fu_o, fu_1, fu_2, rup(DF, BKS),
          abf, qkv, obuf, tmp, ffh, stream);

  // 5) utterance vector: [mean(gt_sen), mean(wav,T), mean(fusion,W)] -> [16,3840]
  {
    dim3 g((SSL + TPB - 1) / TPB, Bv);
    mean_axis1_kernel<<<g, blk, 0, stream>>>(gts, uttr, Bv, Stxt, SSL, 3840, 0);
    mean_axis1_kernel<<<g, blk, 0, stream>>>(wav, uttr, Bv, Tv, SSL, 3840, 768);
  }
  {
    dim3 g((DF + TPB - 1) / TPB, Bv);
    mean_axis1_kernel<<<g, blk, 0, stream>>>(x_fusion, uttr, Bv, Wn, DF, 3840, 1536);
  }

  // 6) sentence heads -> out[0..63] (o-major: sen[:,o] contiguous)
  sen_head_kernel<<<Bv * 4, blk, 0, stream>>>(uttr, sen_w, sen_b, out);

  // 7) word encoder (D=2456, 2 heads): concat(wav_al, gte, asr, af, pv)
  copy_cols_kernel<<<grid_for((size_t)rows * SSL), blk, 0, stream>>>(
      wav_al, x_word, rows, SSL, SSL, DW, 0);
  copy_cols_kernel<<<grid_for((size_t)rows * TXT), blk, 0, stream>>>(
      gte, x_word, rows, TXT, TXT, DW, 768);
  copy_cols_kernel<<<grid_for((size_t)rows * TXT), blk, 0, stream>>>(
      asr, x_word, rows, TXT, TXT, DW, 1536);
  copy_cols_kernel<<<grid_for((size_t)rows * ALN), blk, 0, stream>>>(
      af, x_word, rows, ALN, ALN, DW, 2304);
  copy_cols_kernel<<<grid_for((size_t)rows * PHN), blk, 0, stream>>>(
      pv, x_word, rows, PHN, PHN, DW, 2314);
  run_enc(x_word, DW, 2, word_p, wo_q, wo_o, wo_1, wo_2, rup(DW, BKS),
          abf, qkv, obuf, tmp, ffh, stream);

  // 8) word heads (Conv1d k=1) -> out[64 + o*2048 + b*W + w]
  word_head_kernel<<<rows, blk, 0, stream>>>(x_word, wrd_w, wrd_b, out + 64, DW, rows);
}
